// Contraction_57166014710103
// MI455X (gfx1250) — compile-verified, matrix operand order checked by reading, not run on hardware
//
#include <hip/hip_runtime.h>

typedef __attribute__((ext_vector_type(16))) _Float16 v16h;
typedef __attribute__((ext_vector_type(8)))  float    v8f;

namespace {
constexpr int Bn = 8192, Cn = 128, Ln = 16, En = 10;
constexpr int K3n = 23, K2n = 4;
constexpr int NT3 = 23;         // corr3 N = 368 = 23 tiles of 16 cols (n = q*23 + k)
constexpr int KS3 = 8;          // corr3 K = 256 = 8 wmma ksteps of 32
constexpr int KS2 = 2;          // corr2 K = 64  = 2 wmma ksteps
constexpr int CH3 = NT3 * KS3;  // 184 swizzled B chunks (512 f16 = 1KB each)
constexpr int XSTR = 17, W3STR = 129, W2STR = 129;  // LDS pads (bank-conflict free)
}

// f16 element j of a v16h maps to K-row: lanes 0-15 hold K {0..7,16..23},
// lanes 16-31 hold K {8..15,24..31} (per CDNA5 16-bit WMMA operand layout).
__device__ __forceinline__ int krow16(int lane, int j) {
  return ((j < 8) ? j : j + 8) + ((lane < 16) ? 0 : 8);
}

// Pre-swizzle U3 (as (p,i) x (q,k) matrix) and U2 (as (p,k2) x q) into f16
// chunks laid out so each lane's B operand is one contiguous 32-byte load.
__global__ void prep_swizzle(const float* __restrict__ U3,
                             const float* __restrict__ U2,
                             _Float16* __restrict__ ws3,
                             _Float16* __restrict__ ws2) {
  int tid = blockIdx.x * blockDim.x + threadIdx.x;
  if (tid >= (CH3 + KS2) * 512) return;
  int chunk = tid >> 9;
  int slot  = tid & 511;
  int lane  = slot >> 4;
  int j     = slot & 15;
  if (chunk < CH3) {
    int t = chunk >> 3, s = chunk & 7;
    int n = t * 16 + (lane & 15);
    int q = n / K3n, k = n - q * K3n;
    int kk = s * 32 + krow16(lane, j);        // (p,i) flattened K-row
    int p = kk >> 4, i = kk & 15;
    ws3[(size_t)chunk * 512 + slot] =
        (_Float16)U3[(((p * 16 + q) * 16) + i) * K3n + k];
  } else {
    int s = chunk - CH3;
    int q = lane & 15;
    int kk = s * 32 + krow16(lane, j);        // (p,k2) flattened K-row
    int p = kk >> 2, k2 = kk & 3;
    ws2[(size_t)s * 512 + slot] = (_Float16)U2[(p * 16 + q) * 4 + k2];
  }
}

__global__ __launch_bounds__(256)
void mace_sc_kernel(const float* __restrict__ x,
                    const float* __restrict__ y,
                    const float* __restrict__ U1,
                    const float* __restrict__ W1,
                    const float* __restrict__ W2,
                    const float* __restrict__ W3,
                    const _Float16* __restrict__ ws3,
                    const _Float16* __restrict__ ws2,
                    float* __restrict__ out) {
  __shared__ float ys[En];
  __shared__ float u1s[Ln];
  __shared__ float xs[Cn * XSTR];
  __shared__ float w3s[K3n * W3STR];
  __shared__ float w2s[K2n * W2STR];
  __shared__ float w1s[Cn];

  const int b    = blockIdx.x;
  const int tid  = threadIdx.x;
  const int wave = tid >> 5;
  const int lane = tid & 31;

  if (tid < En) ys[tid] = y[b * En + tid];
  if (tid >= 32 && tid < 32 + Ln) u1s[tid - 32] = U1[tid - 32];
  __syncthreads();

  // Element-mixed weights W~n[k,c] = sum_e y[b,e] * Wn[e,k,c] (kept in LDS;
  // recomputing per block beats 192MB of HBM round-trip for a W~3 tensor).
  for (int o = tid; o < K3n * Cn; o += 256) {
    int k = o >> 7, c = o & (Cn - 1);
    float a = 0.f;
#pragma unroll
    for (int e = 0; e < En; ++e) a += ys[e] * W3[(e * K3n + k) * Cn + c];
    w3s[k * W3STR + c] = a;
  }
  for (int o = tid; o < K2n * Cn; o += 256) {
    int k = o >> 7, c = o & (Cn - 1);
    float a = 0.f;
#pragma unroll
    for (int e = 0; e < En; ++e) a += ys[e] * W2[(e * K2n + k) * Cn + c];
    w2s[k * W2STR + c] = a;
  }
  if (tid < Cn) {
    float a = 0.f;
#pragma unroll
    for (int e = 0; e < En; ++e) a += ys[e] * W1[e * Cn + tid];
    w1s[tid] = a;
  }
  for (int o = tid; o < Cn * Ln; o += 256) {
    int c = o >> 4, i = o & 15;
    xs[c * XSTR + i] = x[((size_t)b * Cn + c) * Ln + i];
  }
  __syncthreads();

  // ---- per-wave WMMA work: A rows m = 0..15 <-> channels c = wave*16 + m ----
  const int mA = lane & 15;
  const int cA = (wave << 4) + mA;
  float xr[Ln];
#pragma unroll
  for (int i = 0; i < Ln; ++i) xr[i] = xs[cA * XSTR + i];

  // Hoisted corr-3 A operands: A[m,(p,i)] = x_p * x_i  (t-loop invariant)
  v16h A3[KS3];
#pragma unroll
  for (int s = 0; s < KS3; ++s) {
#pragma unroll
    for (int j = 0; j < 16; ++j) {
      int kk = s * 32 + krow16(lane, j);
      A3[s][j] = (_Float16)(xr[kk >> 4] * xr[kk & 15]);
    }
  }

  const int mbase = (lane < 16) ? 0 : 8;  // C/D row group of this lane
  float facc[8];
#pragma unroll
  for (int r = 0; r < 8; ++r) facc[r] = 0.f;

  // corr-3: H[m,(q,k)] = sum_{(p,i)} (x_p x_i) * U3[p,q,i,k], then
  // contribution = sum_{(q,k)} H * x_q * W~3[k]   (epilogue fused per tile)
  for (int t = 0; t < NT3; ++t) {
    v8f acc = {};
#pragma unroll
    for (int s = 0; s < KS3; ++s) {
      v16h bmat = *(const v16h*)(ws3 + ((size_t)(t * KS3 + s) * 512 + lane * 16));
      acc = __builtin_amdgcn_wmma_f32_16x16x32_f16(false, A3[s], false, bmat,
                                                   (short)0, acc, false, false);
    }
    int n = t * 16 + (lane & 15);
    int q = n / K3n, k = n - q * K3n;
#pragma unroll
    for (int r = 0; r < 8; ++r) {
      int c = (wave << 4) + mbase + r;
      facc[r] += acc[r] * xs[c * XSTR + q] * w3s[k * W3STR + c];
    }
  }

  // corr-2: A2[m,(p,k2)] = x_p * W~2[k2];  B2[(p,k2),q] = U2[p,q,k2]
  {
    float w2r[K2n];
#pragma unroll
    for (int k = 0; k < K2n; ++k) w2r[k] = w2s[k * W2STR + cA];
    v8f acc = {};
#pragma unroll
    for (int s = 0; s < KS2; ++s) {
      v16h a2;
#pragma unroll
      for (int j = 0; j < 16; ++j) {
        int kk = s * 32 + krow16(lane, j);
        a2[j] = (_Float16)(xr[kk >> 2] * w2r[kk & 3]);
      }
      v16h bmat = *(const v16h*)(ws2 + ((size_t)s * 512 + lane * 16));
      acc = __builtin_amdgcn_wmma_f32_16x16x32_f16(false, a2, false, bmat,
                                                   (short)0, acc, false, false);
    }
    int q = lane & 15;
#pragma unroll
    for (int r = 0; r < 8; ++r) {
      int c = (wave << 4) + mbase + r;
      facc[r] += acc[r] * xs[c * XSTR + q];
    }
  }

  // Reduce the 16 column-lanes inside each half-wave (stays within halves).
#pragma unroll
  for (int m = 1; m <= 8; m <<= 1) {
#pragma unroll
    for (int r = 0; r < 8; ++r) facc[r] += __shfl_xor(facc[r], m, 32);
  }

  // corr-1 + writeout: lanes 0 and 16 own rows 0..7 / 8..15 of their wave.
  if ((lane & 15) == 0) {
#pragma unroll
    for (int r = 0; r < 8; ++r) {
      int c = (wave << 4) + mbase + r;
      float s1 = 0.f;
#pragma unroll
      for (int i = 0; i < Ln; ++i) s1 += u1s[i] * xs[c * XSTR + i];
      out[(size_t)b * Cn + c] = facc[r] + w1s[c] * s1;
    }
  }
}

extern "C" void kernel_launch(void* const* d_in, const int* in_sizes, int n_in,
                              void* d_out, int out_size, void* d_ws, size_t ws_size,
                              hipStream_t stream) {
  (void)in_sizes; (void)n_in; (void)out_size; (void)ws_size;
  const float* x  = (const float*)d_in[0];
  const float* y  = (const float*)d_in[1];
  const float* U1 = (const float*)d_in[2];
  const float* U2 = (const float*)d_in[3];
  const float* U3 = (const float*)d_in[4];
  const float* W1 = (const float*)d_in[5];
  const float* W2 = (const float*)d_in[6];
  const float* W3 = (const float*)d_in[7];

  _Float16* ws3 = (_Float16*)d_ws;            // 184 KB swizzled f16 U3
  _Float16* ws2 = ws3 + (size_t)CH3 * 512;    // + 2 KB swizzled f16 U2

  int prepTotal = (CH3 + KS2) * 512;
  prep_swizzle<<<(prepTotal + 255) / 256, 256, 0, stream>>>(U3, U2, ws3, ws2);
  mace_sc_kernel<<<Bn, 256, 0, stream>>>(x, y, U1, W1, W2, W3, ws3, ws2,
                                         (float*)d_out);
}
